// IBRNet_24859270709709
// MI455X (gfx1250) — matrix operand order repaired and checked by reading, not compiled, not run on hardware
//
#include <hip/hip_runtime.h>

// ---------------------------------------------------------------------------
// IBRNet forward, fused, for gfx1250 (CDNA5, wave32, WMMA f16->f32).
//
//   pack_kernel  : f32 weights -> f16 WMMA B-fragment layout in d_ws
//   point_kernel : 1 wave per 2 sample points (16 view-rows = one WMMA M tile).
//                  ALL MLPs (dirf, base, vis, geo, rgb) on v_wmma_f32_16x16x32_f16;
//                  only cross-view reductions / elementwise glue on VALU.
//                  Wave-entry global_prefetch of the rgb_feat/ray_diff block
//                  overlaps HBM latency with the rd-MLP WMMA phase.
//   attn_kernel  : 1 block (2 waves) per ray. q/k/v, wo(+residual), og0 via WMMA;
//                  D_K=4 attention core + layernorm on VALU (WMMA K=32 would be
//                  8x padding waste and needs cross-lane softmax).
// ---------------------------------------------------------------------------

typedef _Float16 v16h __attribute__((ext_vector_type(16)));
typedef _Float16 v8h  __attribute__((ext_vector_type(8)));
typedef float    v8f  __attribute__((ext_vector_type(8)));

__device__ __forceinline__ float elu_f(float x)  { return x > 0.f ? x : __expf(x) - 1.f; }
__device__ __forceinline__ float sigm_f(float x) { return 1.f / (1.f + __expf(-x)); }

// A fragment (16x32 f16): lane<16 holds row=lane, K = {k0..k0+7, k0+16..k0+23};
// lane>=16 holds row=lane-16, K = {k0+8..k0+15, k0+24..k0+31}.
__device__ __forceinline__ v16h ldA(const _Float16* buf, int strideH, int k0, int lane) {
  int row = lane & 15;
  int off = row * strideH + k0 + ((lane & 16) ? 8 : 0);
  union { v16h v; v8h h[2]; } u;
  u.h[0] = *(const v8h*)(buf + off);
  u.h[1] = *(const v8h*)(buf + off + 16);
  return u.v;
}

// B fragment: packed contiguously, 16 halfs per lane.
__device__ __forceinline__ v16h ldB(const _Float16* __restrict__ g, int lane) {
  const _Float16* p = g + lane * 16;
  union { v16h v; v8h h[2]; } u;
  u.h[0] = *(const v8h*)(p);
  u.h[1] = *(const v8h*)(p + 8);
  return u.v;
}

__device__ __forceinline__ v8f wmma_f16(v16h a, v16h b, v8f c) {
  return __builtin_amdgcn_wmma_f32_16x16x32_f16(false, a, false, b, (short)0, c, false, false);
}

// One 16x16 output tile of X @ W + b: nK k-steps of 32, tile column nt of nN.
__device__ __forceinline__ v8f mm_tile(const _Float16* in, int inStride, int nK,
                                       const _Float16* __restrict__ wpack, int nN, int nt,
                                       float bv, int lane) {
  v8f acc = {bv, bv, bv, bv, bv, bv, bv, bv};
  for (int kt = 0; kt < nK; ++kt) {
    v16h a = ldA(in, inStride, kt * 32, lane);
    v16h b = ldB(wpack + (kt * nN + nt) * 512, lane);
    acc = wmma_f16(a, b, acc);
  }
  return acc;
}

// --------------------------- weight packing --------------------------------
__global__ __launch_bounds__(32) void pack_kernel(const float* __restrict__ w,
                                                  _Float16* __restrict__ dst,
                                                  int din, int dout, int ntiles) {
  int lane = threadIdx.x;
  int tile = blockIdx.x;
  int kt = tile / ntiles, nt = tile % ntiles;
  int n  = nt * 16 + (lane & 15);
  int k0 = kt * 32 + ((lane & 16) ? 8 : 0);
  _Float16* o = dst + tile * 512 + lane * 16;
#pragma unroll
  for (int i = 0; i < 16; ++i) {
    int k = k0 + i + ((i >= 8) ? 8 : 0);
    float v = (k < din && n < dout) ? w[k * dout + n] : 0.f;
    o[i] = (_Float16)v;
  }
}

// Packed weight offsets (halfs):
#define WP_BASE0  0       // 4x4 tiles
#define WP_BASE1  8192    // 2x2
#define WP_VIS0   10240   // 1x2
#define WP_VIS1   11264   // 1x3
#define WP_VIS2A  12800   // 1x2
#define WP_RGB0   13824   // 2x1
#define WP_RGB1   14848   // 1x1
#define WP_RD0    15360   // 1x1
#define WP_RD1    15872   // 1x3
#define WP_GEO0   17408   // 3x4
#define WP_GEO1   23552   // 2x1
#define WP_WQ     24576   // 1x1
#define WP_WK     25088   // 1x1
#define WP_WV     25600   // 1x1
#define WP_WO     26112   // 1x1
#define WP_OG0    26624   // 1x1

// --------------------------- per-point kernel ------------------------------
__global__ __launch_bounds__(32) void point_kernel(
    const float* __restrict__ rgb_feat, const float* __restrict__ ray_diff,
    const float* __restrict__ maskp, const float* __restrict__ sptr,
    const float* __restrict__ rd0b, const float* __restrict__ rd1b,
    const float* __restrict__ b0b,  const float* __restrict__ b1b,
    const float* __restrict__ v0b,  const float* __restrict__ v1b,
    const float* __restrict__ v2ab, const float* __restrict__ v2bw,
    const float* __restrict__ v2bb,
    const float* __restrict__ g0b,  const float* __restrict__ g1b,
    const float* __restrict__ r0b,  const float* __restrict__ r1b,
    const float* __restrict__ r2w,  const float* __restrict__ r2b,
    const _Float16* __restrict__ Wp,
    float* __restrict__ gG, float* __restrict__ gNV, float* __restrict__ out) {

  __shared__ __align__(16) _Float16 sA[16][136];  // matmul A staging (<=128 wide)
  __shared__ __align__(16) _Float16 sB[16][72];   // hidden f16 (<=64 wide)
  __shared__ float sFeat[16][36];
  __shared__ float sX[16][36];
  __shared__ float sRd[16][4];
  __shared__ float sRgbIn[16][3];
  __shared__ float sH16[16][16];
  __shared__ float sE[16], sW[16], sM[16], sVis[16], sW2[16], sRow[16];
  __shared__ float sMean[2][36], sVar[2][36];
  __shared__ float sG65[2][68];
  __shared__ float sRed[2][4];

  const int lane = threadIdx.x;
  const long long P0 = (long long)blockIdx.x * 2;  // first of 2 sample points
  const long long rowG0 = P0 * 8;                  // first global view-row
  const float sval = fabsf(sptr[0]);

  // ---- prefetch this wave's input block: rgb_feat 2240 B (18 lines of 128 B),
  //      ray_diff 256 B (2 lines). global_prefetch_b8, no counters, fire+forget.
  {
    const char* pf = (const char*)(rgb_feat + rowG0 * 35);
    if (lane < 18) __builtin_prefetch(pf + lane * 128, 0, 3);
    const char* pr = (const char*)(ray_diff + rowG0 * 4);
    if (lane >= 18 && lane < 20) __builtin_prefetch(pr + (lane - 18) * 128, 0, 3);
  }

  // ---- phase 0: load ray_diff/mask, e-weights, rd A-staging ----
  for (int idx = lane; idx < 16 * 4; idx += 32) {
    int r_ = idx >> 2, c = idx & 3;
    sRd[r_][c] = ray_diff[(rowG0 + r_) * 4 + c];
  }
  __syncthreads();
  if (lane < 16) {
    sM[lane] = maskp[rowG0 + lane];
    sE[lane] = __expf(sval * (sRd[lane][3] - 1.f));
  }
  for (int idx = lane; idx < 16 * 32; idx += 32) {   // rd0 A: [rd | 0]
    int r_ = idx >> 5, f = idx & 31;
    sA[r_][f] = (_Float16)((f < 4) ? sRd[r_][f] : 0.f);
  }
  for (int idx = lane; idx < 16 * 23; idx += 32) {   // zero x105 K-pad cols
    int r_ = idx / 23, f = idx % 23;
    sA[r_][105 + f] = (_Float16)0.f;
  }
  __syncthreads();
  // ---- rd0: 4->16, elu (WMMA) ----
  {
    int col = lane & 15;
    v8f acc = mm_tile(&sA[0][0], 136, 1, Wp + WP_RD0, 1, 0, rd0b[col], lane);
    int rb = (lane & 16) ? 8 : 0;
#pragma unroll
    for (int i = 0; i < 8; ++i) sB[rb + i][col] = (_Float16)elu_f(acc[i]);
  }
  for (int idx = lane; idx < 16 * 16; idx += 32) {   // zero K-pad for rd1
    int r_ = idx >> 4, f = idx & 15;
    sB[r_][16 + f] = (_Float16)0.f;
  }
  __syncthreads();
  // ---- rd1: 16->35, dirf; feat = rgb_feat + elu(dirf) (WMMA, fused D-store) ----
  for (int nt = 0; nt < 3; ++nt) {
    int col = nt * 16 + (lane & 15);
    float bv = (col < 35) ? rd1b[col] : 0.f;
    v8f acc = mm_tile(&sB[0][0], 72, 1, Wp + WP_RD1, 3, nt, bv, lane);
    if (col < 35) {
      int rb = (lane & 16) ? 8 : 0;
#pragma unroll
      for (int i = 0; i < 8; ++i) {
        int row = rb + i;
        float rf = rgb_feat[(rowG0 + row) * 35 + col];   // coalesced per i
        float fe = rf + elu_f(acc[i]);
        sFeat[row][col] = fe;
        sA[row][70 + col] = (_Float16)fe;                // feat part of x105
        if (col < 3) sRgbIn[row][col] = rf;
      }
    }
  }
  __syncthreads();
  // ---- per-point view weights w ----
  if (lane < 2) {
    float mn = 1e30f;
    for (int v = 0; v < 8; ++v) mn = fminf(mn, sE[lane * 8 + v]);
    float sum = 0.f;
    for (int v = 0; v < 8; ++v) {
      float w = (sE[lane * 8 + v] - mn) * sM[lane * 8 + v];
      sW[lane * 8 + v] = w; sum += w;
    }
    sRed[lane][0] = sum + 1e-8f;
  }
  __syncthreads();
  if (lane < 16) sW[lane] = sW[lane] / sRed[lane >> 3][0];
  __syncthreads();
  // ---- weighted mean / var over views -> x105 mean|var part ----
  for (int idx = lane; idx < 2 * 35; idx += 32) {
    int p = idx / 35, f = idx % 35;
    float m = 0.f;
    for (int v = 0; v < 8; ++v) m += sW[p * 8 + v] * sFeat[p * 8 + v][f];
    float vv = 0.f;
    for (int v = 0; v < 8; ++v) { float d = sFeat[p * 8 + v][f] - m; vv += sW[p * 8 + v] * d * d; }
    sMean[p][f] = m; sVar[p][f] = vv;
  }
  __syncthreads();
  for (int idx = lane; idx < 16 * 70; idx += 32) {
    int r_ = idx / 70, f = idx % 70, p = r_ >> 3;
    sA[r_][f] = (_Float16)((f < 35) ? sMean[p][f] : sVar[p][f - 35]);
  }
  __syncthreads();
  // ---- base0: 105->64, elu ----
  for (int nt = 0; nt < 4; ++nt) {
    int col = nt * 16 + (lane & 15);
    v8f acc = mm_tile(&sA[0][0], 136, 4, Wp + WP_BASE0, 4, nt, b0b[col], lane);
    int rb = (lane & 16) ? 8 : 0;
#pragma unroll
    for (int i = 0; i < 8; ++i) sB[rb + i][col] = (_Float16)elu_f(acc[i]);
  }
  __syncthreads();
  // ---- base1: 64->32, elu -> x (f32) ----
  for (int nt = 0; nt < 2; ++nt) {
    int col = nt * 16 + (lane & 15);
    v8f acc = mm_tile(&sB[0][0], 72, 2, Wp + WP_BASE1, 2, nt, b1b[col], lane);
    int rb = (lane & 16) ? 8 : 0;
#pragma unroll
    for (int i = 0; i < 8; ++i) sX[rb + i][col] = elu_f(acc[i]);
  }
  __syncthreads();
  // ---- vis0(x*w) -> vis1 -> x_res, vis ----
  for (int idx = lane; idx < 16 * 32; idx += 32) {
    int r_ = idx >> 5, f = idx & 31;
    sA[r_][f] = (_Float16)(sX[r_][f] * sW[r_]);
  }
  __syncthreads();
  for (int nt = 0; nt < 2; ++nt) {
    int col = nt * 16 + (lane & 15);
    v8f acc = mm_tile(&sA[0][0], 136, 1, Wp + WP_VIS0, 2, nt, v0b[col], lane);
    int rb = (lane & 16) ? 8 : 0;
#pragma unroll
    for (int i = 0; i < 8; ++i) sB[rb + i][col] = (_Float16)elu_f(acc[i]);
  }
  __syncthreads();
  for (int nt = 0; nt < 3; ++nt) {   // vis1: 32->33
    int col = nt * 16 + (lane & 15);
    float bv = (col < 33) ? v1b[col] : 0.f;
    v8f acc = mm_tile(&sB[0][0], 72, 1, Wp + WP_VIS1, 3, nt, bv, lane);
    int rb = (lane & 16) ? 8 : 0;
    if (col < 32) {
#pragma unroll
      for (int i = 0; i < 8; ++i) sX[rb + i][col] += elu_f(acc[i]);   // x += x_res
    } else if (col == 32) {
#pragma unroll
      for (int i = 0; i < 8; ++i) sRow[rb + i] = elu_f(acc[i]);       // xv[...,32]
    }
  }
  __syncthreads();
  if (lane < 16) sVis[lane] = sigm_f(sRow[lane]) * sM[lane];
  __syncthreads();
  // ---- vis2a(x*vis) -> vis2b -> vis (final) ----
  for (int idx = lane; idx < 16 * 32; idx += 32) {
    int r_ = idx >> 5, f = idx & 31;
    sA[r_][f] = (_Float16)(sX[r_][f] * sVis[r_]);
  }
  __syncthreads();
  for (int nt = 0; nt < 2; ++nt) {
    int col = nt * 16 + (lane & 15);
    v8f acc = mm_tile(&sA[0][0], 136, 1, Wp + WP_VIS2A, 2, nt, v2ab[col], lane);
    int rb = (lane & 16) ? 8 : 0;
#pragma unroll
    for (int i = 0; i < 8; ++i) sB[rb + i][col] = (_Float16)elu_f(acc[i]);
  }
  __syncthreads();
  if (lane < 16) {
    float a = v2bb[0];
#pragma unroll
    for (int j = 0; j < 32; ++j) a += (float)sB[lane][j] * v2bw[j];
    sVis[lane] = sigm_f(a) * sM[lane];
  }
  __syncthreads();
  if (lane < 2) {
    float s = 0.f;
    for (int v = 0; v < 8; ++v) s += sVis[lane * 8 + v];
    sRed[lane][0] = s + 1e-8f;
  }
  __syncthreads();
  if (lane < 16) sW2[lane] = sVis[lane] / sRed[lane >> 3][0];
  __syncthreads();
  // ---- mean2/var2 over x, glob65 = [mean2 | var2 | mean(w2)] ----
  for (int idx = lane; idx < 2 * 32; idx += 32) {
    int p = idx >> 5, f = idx & 31;
    float m = 0.f;
    for (int v = 0; v < 8; ++v) m += sW2[p * 8 + v] * sX[p * 8 + v][f];
    float vv = 0.f;
    for (int v = 0; v < 8; ++v) { float d = sX[p * 8 + v][f] - m; vv += sW2[p * 8 + v] * d * d; }
    sG65[p][f] = m; sG65[p][32 + f] = vv;
  }
  if (lane < 2) {
    float s = 0.f, nvs = 0.f;
    for (int v = 0; v < 8; ++v) { s += sW2[lane * 8 + v]; nvs += sM[lane * 8 + v]; }
    sG65[lane][64] = s * 0.125f;
    sRed[lane][1] = nvs;
  }
  __syncthreads();
  // ---- geo0 (65->64) / geo1 (64->16) via WMMA (rows 0-1 valid, rest waste) ----
  for (int idx = lane; idx < 2 * 96; idx += 32) {
    int p = idx / 96, k = idx % 96;
    sA[p][k] = (_Float16)((k < 65) ? sG65[p][k] : 0.f);
  }
  __syncthreads();
  for (int nt = 0; nt < 4; ++nt) {
    int col = nt * 16 + (lane & 15);
    v8f acc = mm_tile(&sA[0][0], 136, 3, Wp + WP_GEO0, 4, nt, g0b[col], lane);
    if (lane < 16) {
#pragma unroll
      for (int i = 0; i < 2; ++i) sB[i][col] = (_Float16)elu_f(acc[i]);
    }
  }
  __syncthreads();
  {
    int col = lane & 15;
    v8f acc = mm_tile(&sB[0][0], 72, 2, Wp + WP_GEO1, 1, 0, g1b[col], lane);
    if (lane < 16) {
#pragma unroll
      for (int p = 0; p < 2; ++p) {
        float a = elu_f(acc[p]);
        long long P = P0 + p;
        int sidx = (int)(P & 63);
        float expo = (float)(2 * (col >> 1)) * (1.f / 16.f);
        float angle = (float)sidx * __expf(-expo * 9.210340371976184f); // ln(1e4)
        a += (col & 1) ? cosf(angle) : sinf(angle);
        gG[P * 16 + col] = a;
        if (col == 0) gNV[P] = sRed[p][1];
      }
    }
  }
  __syncthreads();
  // ---- rgb branch: xr = [x | vis | ray_diff] -> 16 -> 8 -> 1, blend ----
  for (int idx = lane; idx < 16 * 64; idx += 32) {
    int r_ = idx >> 6, f = idx & 63;
    float v;
    if (f < 32)       v = sX[r_][f];
    else if (f == 32) v = sVis[r_];
    else if (f < 37)  v = sRd[r_][f - 33];
    else              v = 0.f;
    sA[r_][f] = (_Float16)v;
  }
  __syncthreads();
  {
    int col = lane & 15;
    v8f acc = mm_tile(&sA[0][0], 136, 2, Wp + WP_RGB0, 1, 0, r0b[col], lane);
    int rb = (lane & 16) ? 8 : 0;
#pragma unroll
    for (int i = 0; i < 8; ++i) sB[rb + i][col] = (_Float16)elu_f(acc[i]);
  }
  for (int idx = lane; idx < 16 * 16; idx += 32) {  // zero K-pad cols for rgb1
    int r_ = idx >> 4, f = idx & 15;
    sB[r_][16 + f] = (_Float16)0.f;
  }
  __syncthreads();
  {
    int col = lane & 15;
    float bv = (col < 8) ? r1b[col] : 0.f;
    v8f acc = mm_tile(&sB[0][0], 72, 1, Wp + WP_RGB1, 1, 0, bv, lane);
    int rb = (lane & 16) ? 8 : 0;
    if (col < 8)
#pragma unroll
      for (int i = 0; i < 8; ++i) sH16[rb + i][col] = elu_f(acc[i]);
  }
  __syncthreads();
  if (lane < 16) {
    float a = r2b[0];
#pragma unroll
    for (int j = 0; j < 8; ++j) a += sH16[lane][j] * r2w[j];
    sRow[lane] = (sM[lane] == 0.f) ? -1e9f : a;
  }
  __syncthreads();
  if (lane < 2) {
    float mx = -1e30f;
    for (int v = 0; v < 8; ++v) mx = fmaxf(mx, sRow[lane * 8 + v]);
    float s = 0.f;
    for (int v = 0; v < 8; ++v) { float e = __expf(sRow[lane * 8 + v] - mx); sE[lane * 8 + v] = e; s += e; }
    sRed[lane][2] = s;
  }
  __syncthreads();
  for (int idx = lane; idx < 6; idx += 32) {
    int p = idx / 3, c = idx % 3;
    float inv = 1.f / sRed[p][2];
    float s = 0.f;
    for (int v = 0; v < 8; ++v) s += sRgbIn[p * 8 + v][c] * (sE[p * 8 + v] * inv);
    out[(P0 + p) * 4 + c] = s;
  }
}

// --------------------------- per-ray attention -----------------------------
__global__ __launch_bounds__(64) void attn_kernel(
    const float* __restrict__ gG, const float* __restrict__ gNV,
    const _Float16* __restrict__ Wp,
    const float* __restrict__ lng, const float* __restrict__ lnb,
    const float* __restrict__ og0b,
    const float* __restrict__ og1w, const float* __restrict__ og1b,
    float* __restrict__ out) {
  __shared__ __align__(16) _Float16 sGh[64][40];  // f16 A staging (cols 16-31 zero)
  __shared__ float sg[64][17];
  __shared__ float sq[64][17], sk[64][17], sv[64][17];
  __shared__ float sS[64][68];
  __shared__ float sNV[64];

  const int t = threadIdx.x;
  const int lane = t & 31;
  const int wave = t >> 5;
  const long long ray = blockIdx.x;
  const long long tokBase = ray * 64;

  __builtin_prefetch((const char*)(gG + tokBase * 16) + t * 64, 0, 3);

#pragma unroll
  for (int j = 0; j < 16; ++j) {
    float v = gG[(tokBase + t) * 16 + j];
    sg[t][j] = v; sGh[t][j] = (_Float16)v;
  }
#pragma unroll
  for (int j = 16; j < 32; ++j) sGh[t][j] = (_Float16)0.f;
  sNV[t] = gNV[tokBase + t];
  __syncthreads();

  const v8f z8 = {0.f, 0.f, 0.f, 0.f, 0.f, 0.f, 0.f, 0.f};
  // ---- q/k/v projections via WMMA (each wave owns 2 row-tiles) ----
  for (int rt = wave * 2; rt < wave * 2 + 2; ++rt) {
    v16h a = ldA(&sGh[rt * 16][0], 40, 0, lane);
    v8f aq = wmma_f16(a, ldB(Wp + WP_WQ, lane), z8);
    v8f ak = wmma_f16(a, ldB(Wp + WP_WK, lane), z8);
    v8f av = wmma_f16(a, ldB(Wp + WP_WV, lane), z8);
    int col = lane & 15, rb = (lane & 16) ? 8 : 0;
#pragma unroll
    for (int i = 0; i < 8; ++i) {
      int row = rt * 16 + rb + i;
      sq[row][col] = aq[i]; sk[row][col] = ak[i]; sv[row][col] = av[i];
    }
  }
  __syncthreads();

  // ---- D_K=4 attention core (VALU), scores cached in LDS ----
  float o[16];
  float nvq = sNV[t];
  bool masked = !(nvq > 1.f);
  for (int h = 0; h < 4; ++h) {
    float q0 = sq[t][h * 4], q1 = sq[t][h * 4 + 1], q2 = sq[t][h * 4 + 2], q3 = sq[t][h * 4 + 3];
    float mx = -1e30f;
    for (int kk = 0; kk < 64; ++kk) {
      float a = (q0 * sk[kk][h * 4] + q1 * sk[kk][h * 4 + 1] +
                 q2 * sk[kk][h * 4 + 2] + q3 * sk[kk][h * 4 + 3]) * 0.5f;
      if (masked) a = 0.f;
      sS[t][kk] = a;
      mx = fmaxf(mx, a);
    }
    float s = 0.f, a0 = 0.f, a1 = 0.f, a2 = 0.f, a3 = 0.f;
    for (int kk = 0; kk < 64; ++kk) {
      float e = __expf(sS[t][kk] - mx);
      s += e;
      a0 += e * sv[kk][h * 4];     a1 += e * sv[kk][h * 4 + 1];
      a2 += e * sv[kk][h * 4 + 2]; a3 += e * sv[kk][h * 4 + 3];
    }
    float inv = 1.f / s;
    o[h * 4] = a0 * inv; o[h * 4 + 1] = a1 * inv;
    o[h * 4 + 2] = a2 * inv; o[h * 4 + 3] = a3 * inv;
  }
  __syncthreads();                    // all reads of sq/sGh done
#pragma unroll
  for (int j = 0; j < 16; ++j) sGh[t][j] = (_Float16)o[j];
  __syncthreads();
  // ---- o @ wo + glob (WMMA, residual fused into D-store) -> sq ----
  for (int rt = wave * 2; rt < wave * 2 + 2; ++rt) {
    v16h a = ldA(&sGh[rt * 16][0], 40, 0, lane);
    v8f acc = wmma_f16(a, ldB(Wp + WP_WO, lane), z8);
    int col = lane & 15, rb = (lane & 16) ? 8 : 0;
#pragma unroll
    for (int i = 0; i < 8; ++i) {
      int row = rt * 16 + rb + i;
      sq[row][col] = acc[i] + sg[row][col];
    }
  }
  __syncthreads();
  // ---- layernorm (VALU per token) -> sGh ----
  {
    float mu = 0.f;
#pragma unroll
    for (int j = 0; j < 16; ++j) mu += sq[t][j];
    mu *= (1.f / 16.f);
    float var = 0.f;
#pragma unroll
    for (int j = 0; j < 16; ++j) { float d = sq[t][j] - mu; var += d * d; }
    var *= (1.f / 16.f);
    float rstd = rsqrtf(var + 1e-6f);
#pragma unroll
    for (int j = 0; j < 16; ++j)
      sGh[t][j] = (_Float16)((sq[t][j] - mu) * rstd * lng[j] + lnb[j]);
  }
  __syncthreads();
  // ---- og0: 16->16, elu (WMMA) -> sk ----
  for (int rt = wave * 2; rt < wave * 2 + 2; ++rt) {
    int col = lane & 15;
    float bv = og0b[col];
    v8f acc = {bv, bv, bv, bv, bv, bv, bv, bv};
    v16h a = ldA(&sGh[rt * 16][0], 40, 0, lane);
    acc = wmma_f16(a, ldB(Wp + WP_OG0, lane), acc);
    int rb = (lane & 16) ? 8 : 0;
#pragma unroll
    for (int i = 0; i < 8; ++i) sk[rt * 16 + rb + i][col] = elu_f(acc[i]);
  }
  __syncthreads();
  // ---- og1: 16->1, relu, mask ----
  {
    float a = og1b[0];
#pragma unroll
    for (int i = 0; i < 16; ++i) a += sk[t][i] * og1w[i];
    float sigma = fmaxf(a, 0.f);
    if (nvq < 1.f) sigma = 0.f;
    out[(tokBase + t) * 4 + 3] = sigma;
  }
}

// --------------------------- launch ----------------------------------------
extern "C" void kernel_launch(void* const* d_in, const int* in_sizes, int n_in,
                              void* d_out, int out_size, void* d_ws, size_t ws_size,
                              hipStream_t stream) {
  const float* rgb_feat = (const float*)d_in[0];
  const float* ray_diff = (const float*)d_in[1];
  const float* maskp    = (const float*)d_in[2];
  const float* sptr     = (const float*)d_in[3];
  const float* rd0w = (const float*)d_in[4];  const float* rd0b = (const float*)d_in[5];
  const float* rd1w = (const float*)d_in[6];  const float* rd1b = (const float*)d_in[7];
  const float* b0w  = (const float*)d_in[8];  const float* b0b  = (const float*)d_in[9];
  const float* b1w  = (const float*)d_in[10]; const float* b1b  = (const float*)d_in[11];
  const float* v0w  = (const float*)d_in[12]; const float* v0b  = (const float*)d_in[13];
  const float* v1w  = (const float*)d_in[14]; const float* v1b  = (const float*)d_in[15];
  const float* v2aw = (const float*)d_in[16]; const float* v2ab = (const float*)d_in[17];
  const float* v2bw = (const float*)d_in[18]; const float* v2bb = (const float*)d_in[19];
  const float* g0w  = (const float*)d_in[20]; const float* g0b  = (const float*)d_in[21];
  const float* g1w  = (const float*)d_in[22]; const float* g1b  = (const float*)d_in[23];
  const float* wq   = (const float*)d_in[24]; const float* wk   = (const float*)d_in[25];
  const float* wv   = (const float*)d_in[26]; const float* wo   = (const float*)d_in[27];
  const float* lng  = (const float*)d_in[28]; const float* lnb  = (const float*)d_in[29];
  const float* og0w = (const float*)d_in[30]; const float* og0b = (const float*)d_in[31];
  const float* og1w = (const float*)d_in[32]; const float* og1b = (const float*)d_in[33];
  const float* r0w  = (const float*)d_in[34]; const float* r0b  = (const float*)d_in[35];
  const float* r1w  = (const float*)d_in[36]; const float* r1b  = (const float*)d_in[37];
  const float* r2w  = (const float*)d_in[38]; const float* r2b  = (const float*)d_in[39];

  _Float16* Wp = (_Float16*)d_ws;                               // ~54 KiB packed weights
  float* gG  = (float*)((char*)d_ws + 65536);                   // 131072 x 16 f32
  float* gNV = (float*)((char*)d_ws + 65536 + 8388608);         // 131072 f32

  pack_kernel<<<16, 32, 0, stream>>>(b0w,  Wp + WP_BASE0, 105, 64, 4);
  pack_kernel<<< 4, 32, 0, stream>>>(b1w,  Wp + WP_BASE1,  64, 32, 2);
  pack_kernel<<< 2, 32, 0, stream>>>(v0w,  Wp + WP_VIS0,   32, 32, 2);
  pack_kernel<<< 3, 32, 0, stream>>>(v1w,  Wp + WP_VIS1,   32, 33, 3);
  pack_kernel<<< 2, 32, 0, stream>>>(v2aw, Wp + WP_VIS2A,  32, 32, 2);
  pack_kernel<<< 2, 32, 0, stream>>>(r0w,  Wp + WP_RGB0,   37, 16, 1);
  pack_kernel<<< 1, 32, 0, stream>>>(r1w,  Wp + WP_RGB1,   16,  8, 1);
  pack_kernel<<< 1, 32, 0, stream>>>(rd0w, Wp + WP_RD0,     4, 16, 1);
  pack_kernel<<< 3, 32, 0, stream>>>(rd1w, Wp + WP_RD1,    16, 35, 3);
  pack_kernel<<<12, 32, 0, stream>>>(g0w,  Wp + WP_GEO0,   65, 64, 4);
  pack_kernel<<< 2, 32, 0, stream>>>(g1w,  Wp + WP_GEO1,   64, 16, 1);
  pack_kernel<<< 1, 32, 0, stream>>>(wq,   Wp + WP_WQ,     16, 16, 1);
  pack_kernel<<< 1, 32, 0, stream>>>(wk,   Wp + WP_WK,     16, 16, 1);
  pack_kernel<<< 1, 32, 0, stream>>>(wv,   Wp + WP_WV,     16, 16, 1);
  pack_kernel<<< 1, 32, 0, stream>>>(wo,   Wp + WP_WO,     16, 16, 1);
  pack_kernel<<< 1, 32, 0, stream>>>(og0w, Wp + WP_OG0,    16, 16, 1);

  point_kernel<<<65536, 32, 0, stream>>>(
      rgb_feat, ray_diff, maskp, sptr,
      rd0b, rd1b, b0b, b1b, v0b, v1b, v2ab, v2bw, v2bb,
      g0b, g1b, r0b, r1b, r2w, r2b,
      Wp, gG, gNV, (float*)d_out);

  attn_kernel<<<2048, 64, 0, stream>>>(
      gG, gNV, Wp, lng, lnb, og0b, og1w, og1b, (float*)d_out);
}
